// Discriminator_23235773071430
// MI455X (gfx1250) — matrix operand order repaired and checked by reading, not compiled, not run on hardware
//
#include <hip/hip_runtime.h>
#include <hip/hip_bf16.h>

// ---------------------------------------------------------------------------
// GINE discriminator for MI455X (gfx1250, wave32, WMMA bf16).
// ---------------------------------------------------------------------------

#define N_NODES  50000
#define N_EDGES  800000
#define N_GRAPHS 256
#define HIDDEN   128
#define BN_EPS   1e-5f

typedef __bf16 v16bf __attribute__((ext_vector_type(16)));
typedef __bf16 v8bf  __attribute__((ext_vector_type(8)));
typedef __bf16 v4bf  __attribute__((ext_vector_type(4)));
typedef float  v8f   __attribute__((ext_vector_type(8)));
typedef int    v4i   __attribute__((vector_size(16)));   // matches builtin param
typedef __attribute__((address_space(3))) v4i* lds_v4i_p;

// CDNA5 async global->LDS copy (bypasses VGPRs, tracked by ASYNCcnt)
#if defined(__has_builtin)
#if __has_builtin(__builtin_amdgcn_global_load_async_to_lds_b128) && \
    __has_builtin(__builtin_amdgcn_s_wait_asynccnt)
#define HAS_ASYNC 1
#endif
#endif
#ifndef HAS_ASYNC
#define HAS_ASYNC 0
#endif

__device__ __forceinline__ void stage_b128(const __bf16* __restrict__ g,
                                           __bf16* l) {
#if HAS_ASYNC
    __builtin_amdgcn_global_load_async_to_lds_b128((v4i*)g, (lds_v4i_p)l, 0, 0);
#else
    *(v8bf*)l = *(const v8bf*)g;
#endif
}

__device__ __forceinline__ void stage_fence() {
#if HAS_ASYNC
    __builtin_amdgcn_s_wait_asynccnt(0);
#endif
}

// ---------------------------------------------------------------------------
// zero fill
// ---------------------------------------------------------------------------
__global__ void zero_kernel(float* __restrict__ p, int n) {
    int i = blockIdx.x * blockDim.x + threadIdx.x;
    int st = gridDim.x * blockDim.x;
    for (; i < n; i += st) p[i] = 0.0f;
}

// ---------------------------------------------------------------------------
// Convert a weight matrix W[K x 128] (f32, row-major) into fragment-major
// bf16: wf[kc][nt][lane][j] = W[kc*32 + 16*(lane>=16) + j][nt*16 + lane%16].
// One block (256 threads) per K-chunk; each thread writes 16 contiguous bf16.
// Rows >= din are zero-padded (layer-0 K padding).
// ---------------------------------------------------------------------------
__global__ __launch_bounds__(256)
void wfrag_kernel(const float* __restrict__ W, __bf16* __restrict__ wf, int din) {
    int kc = blockIdx.x;
    int tid = threadIdx.x;
    int nt = tid >> 5, ln = tid & 31;
    int col = nt * 16 + (ln & 15);
    int krow = (ln >> 4) * 16;
    __bf16* out = wf + ((size_t)(kc * 8 + nt) * 32 + ln) * 16;
#pragma unroll
    for (int j = 0; j < 16; ++j) {
        int gk = kc * 32 + krow + j;
        float v = (gk < din) ? W[(size_t)gk * HIDDEN + col] : 0.0f;
        out[j] = (__bf16)v;
    }
}

// ---------------------------------------------------------------------------
// Edge message + scatter, din = 10 (layer 0). One thread per edge.
// ---------------------------------------------------------------------------
__global__ __launch_bounds__(256)
void edge_kernel10(const float* __restrict__ ea, const float* __restrict__ We,
                   const float* __restrict__ be, const float* __restrict__ hin,
                   const int* __restrict__ src, const int* __restrict__ dst,
                   float* __restrict__ agg) {
    int e = blockIdx.x * blockDim.x + threadIdx.x;
    if (e >= N_EDGES) return;
    int s = src[e], d = dst[e];
    float a0 = ea[e * 3 + 0], a1 = ea[e * 3 + 1], a2 = ea[e * 3 + 2];
#pragma unroll
    for (int j = 0; j < 10; ++j) {
        float p = be[j] + a0 * We[j] + a1 * We[10 + j] + a2 * We[20 + j];
        float m = fmaxf(hin[(size_t)s * 10 + j] + p, 0.0f);
        atomicAdd(&agg[(size_t)d * 10 + j], m);
    }
}

// ---------------------------------------------------------------------------
// Edge message + scatter, din = 128. One wave32 per edge, 4 channels/lane.
// ---------------------------------------------------------------------------
__global__ __launch_bounds__(256)
void edge_kernel128(const float* __restrict__ ea, const float* __restrict__ We,
                    const float* __restrict__ be, const float* __restrict__ hin,
                    const int* __restrict__ src, const int* __restrict__ dst,
                    float* __restrict__ agg) {
    int lane = threadIdx.x & 31;
    int gw = (blockIdx.x * blockDim.x + threadIdx.x) >> 5;
    int nw = (gridDim.x * blockDim.x) >> 5;

    const float4* h4 = (const float4*)hin;
    const float4* W4 = (const float4*)We;   // [3][32] float4
    const float4* b4 = (const float4*)be;

    float4 w0 = W4[lane], w1 = W4[32 + lane], w2 = W4[64 + lane], bb = b4[lane];

    for (int e = gw; e < N_EDGES; e += nw) {
        int en = e + nw;
        if (en < N_EDGES)  // prefetch next gathered row (global_prefetch_b8)
            __builtin_prefetch((const void*)(hin + (size_t)src[en] * HIDDEN), 0, 1);

        int s = src[e], d = dst[e];
        float a0 = ea[e * 3 + 0], a1 = ea[e * 3 + 1], a2 = ea[e * 3 + 2];
        float4 hv = h4[(size_t)s * 32 + lane];
        float m0 = fmaxf(hv.x + a0 * w0.x + a1 * w1.x + a2 * w2.x + bb.x, 0.0f);
        float m1 = fmaxf(hv.y + a0 * w0.y + a1 * w1.y + a2 * w2.y + bb.y, 0.0f);
        float m2 = fmaxf(hv.z + a0 * w0.z + a1 * w1.z + a2 * w2.z + bb.z, 0.0f);
        float m3 = fmaxf(hv.w + a0 * w0.w + a1 * w1.w + a2 * w2.w + bb.w, 0.0f);
        float* ag = agg + (size_t)d * HIDDEN + lane * 4;
        atomicAdd(ag + 0, m0);
        atomicAdd(ag + 1, m1);
        atomicAdd(ag + 2, m2);
        atomicAdd(ag + 3, m3);
    }
}

// ---------------------------------------------------------------------------
// Fused node MLP:  h2 = relu(z @ W1 + b1) @ W2 + b2,   z = (1+eps)*hin + agg
// Block = 256 threads = 8 waves; block computes a 128x128 tile.
// Weights arrive pre-packed fragment-major (bf16); each weight chunk staging
// is a contiguous 32B/thread copy, done with async global->LDS when the
// toolchain exposes it. All WMMA operands load as aligned ds_load_b128 pairs.
// ---------------------------------------------------------------------------
#define SA_STRIDE 40     // bf16, multiple of 8 -> 16B-aligned half-fragments
#define SH_STRIDE 136    // bf16, multiple of 8
#define BF_STRIDE 24     // bf16 per lane slot (48 B, multiple of 16 B)

template <int DIN>
__global__ __launch_bounds__(256)
void mlp_wmma_kernel(const float* __restrict__ hin, const float* __restrict__ agg,
                     const __bf16* __restrict__ wf1, const float* __restrict__ b1,
                     const __bf16* __restrict__ wf2, const float* __restrict__ b2,
                     const float* __restrict__ epsp,
                     float* __restrict__ h2, float* __restrict__ bnsum,
                     float* __restrict__ bnsq) {
    constexpr int KCH = (DIN + 31) / 32;             // K-chunks of GEMM1

    __shared__ __bf16 sA[128][SA_STRIDE];            // z K-chunk tile (128x32)
    __shared__ __bf16 sBf[8][32][BF_STRIDE];         // fragment-major weight chunk
    __shared__ __bf16 sH[128][SH_STRIDE];            // relu(z@W1+b1) 128x128
    __shared__ float  sRed[256];

    const int tid = threadIdx.x;
    const int lane = tid & 31;
    const int wave = tid >> 5;
    const int half = lane >> 4;
    const int l15 = lane & 15;
    const int rowBase = blockIdx.x * 128;
    const float eps1 = 1.0f + epsp[0];

    const int s_nt = tid >> 5;
    const int s_ln = tid & 31;

    // ---------------- GEMM1: z @ W1 ----------------
    v8f acc[8] = {};
    for (int kc = 0; kc < KCH; ++kc) {
        // issue weight-chunk copy first so it overlaps z staging
        {
            const __bf16* wsrc = wf1 + ((size_t)kc * 256 + tid) * 16;
            stage_b128(wsrc, &sBf[s_nt][s_ln][0]);
            stage_b128(wsrc + 8, &sBf[s_nt][s_ln][8]);
        }
        if (DIN == 128) {
            // vectorized z staging: 4 x float4 per thread, no K guard
#pragma unroll
            for (int i = 0; i < 4; ++i) {
                int idx = tid + i * 256;             // 0..1023
                int r = idx >> 3, c4 = idx & 7;
                int grow = rowBase + r;
                float4 hv = {0.f, 0.f, 0.f, 0.f}, av = {0.f, 0.f, 0.f, 0.f};
                if (grow < N_NODES) {
                    hv = ((const float4*)hin)[(size_t)grow * 32 + kc * 8 + c4];
                    av = ((const float4*)agg)[(size_t)grow * 32 + kc * 8 + c4];
                }
                v4bf z;
                z[0] = (__bf16)(eps1 * hv.x + av.x);
                z[1] = (__bf16)(eps1 * hv.y + av.y);
                z[2] = (__bf16)(eps1 * hv.z + av.z);
                z[3] = (__bf16)(eps1 * hv.w + av.w);
                *(v4bf*)&sA[r][c4 * 4] = z;
            }
        } else {
#pragma unroll
            for (int i = 0; i < 16; ++i) {
                int idx = tid + i * 256;
                int r = idx >> 5, k = idx & 31;
                int grow = rowBase + r, gk = kc * 32 + k;
                float v = 0.0f;
                if (grow < N_NODES && gk < DIN)
                    v = eps1 * hin[(size_t)grow * DIN + gk] +
                        agg[(size_t)grow * DIN + gk];
                sA[r][k] = (__bf16)v;
            }
        }
        stage_fence();
        __syncthreads();

        // A fragment: two aligned 16B runs per lane
        int mrow = wave * 16 + l15;
        v8bf alo = *(const v8bf*)&sA[mrow][half * 16];
        v8bf ahi = *(const v8bf*)&sA[mrow][half * 16 + 8];
        v16bf a = __builtin_shufflevector(alo, ahi, 0, 1, 2, 3, 4, 5, 6, 7,
                                          8, 9, 10, 11, 12, 13, 14, 15);
#pragma unroll
        for (int nt = 0; nt < 8; ++nt) {
            v8bf blo = *(const v8bf*)&sBf[nt][lane][0];
            v8bf bhi = *(const v8bf*)&sBf[nt][lane][8];
            v16bf b = __builtin_shufflevector(blo, bhi, 0, 1, 2, 3, 4, 5, 6, 7,
                                              8, 9, 10, 11, 12, 13, 14, 15);
            acc[nt] = __builtin_amdgcn_wmma_f32_16x16x32_bf16(
                false, a, false, b, (short)0, acc[nt], false, false);
        }
        __syncthreads();
    }

    // bias + relu -> sH (bf16)
#pragma unroll
    for (int nt = 0; nt < 8; ++nt) {
        int col = nt * 16 + l15;
        float bias = b1[col];
#pragma unroll
        for (int r = 0; r < 8; ++r) {
            int row = wave * 16 + half * 8 + r;
            float v = acc[nt][r] + bias;
            sH[row][col] = (__bf16)(v > 0.0f ? v : 0.0f);
        }
    }
    __syncthreads();

    // ---------------- GEMM2: relu(.) @ W2 ----------------
    v8f acc2[8] = {};
    for (int kc = 0; kc < 4; ++kc) {
        {
            const __bf16* wsrc = wf2 + ((size_t)kc * 256 + tid) * 16;
            stage_b128(wsrc, &sBf[s_nt][s_ln][0]);
            stage_b128(wsrc + 8, &sBf[s_nt][s_ln][8]);
        }
        stage_fence();
        __syncthreads();

        int mrow = wave * 16 + l15;
        v8bf alo = *(const v8bf*)&sH[mrow][kc * 32 + half * 16];
        v8bf ahi = *(const v8bf*)&sH[mrow][kc * 32 + half * 16 + 8];
        v16bf a = __builtin_shufflevector(alo, ahi, 0, 1, 2, 3, 4, 5, 6, 7,
                                          8, 9, 10, 11, 12, 13, 14, 15);
#pragma unroll
        for (int nt = 0; nt < 8; ++nt) {
            v8bf blo = *(const v8bf*)&sBf[nt][lane][0];
            v8bf bhi = *(const v8bf*)&sBf[nt][lane][8];
            v16bf b = __builtin_shufflevector(blo, bhi, 0, 1, 2, 3, 4, 5, 6, 7,
                                              8, 9, 10, 11, 12, 13, 14, 15);
            acc2[nt] = __builtin_amdgcn_wmma_f32_16x16x32_bf16(
                false, a, false, b, (short)0, acc2[nt], false, false);
        }
        __syncthreads();
    }

    // epilogue: + b2, store f32, BN partial sums (ds_add_f32 -> global atomic)
    sRed[tid] = 0.0f;
    __syncthreads();
#pragma unroll
    for (int nt = 0; nt < 8; ++nt) {
        int col = nt * 16 + l15;
        float bias = b2[col];
        float ls = 0.0f, lq = 0.0f;
#pragma unroll
        for (int r = 0; r < 8; ++r) {
            int row = rowBase + wave * 16 + half * 8 + r;
            float v = acc2[nt][r] + bias;
            if (row < N_NODES) {
                h2[(size_t)row * HIDDEN + col] = v;
                ls += v;
                lq += v * v;
            }
        }
        atomicAdd(&sRed[col], ls);
        atomicAdd(&sRed[128 + col], lq);
    }
    __syncthreads();
    if (tid < 128) {
        atomicAdd(&bnsum[tid], sRed[tid]);
        atomicAdd(&bnsq[tid], sRed[128 + tid]);
    }
}

// ---------------------------------------------------------------------------
// BatchNorm (training stats) + ReLU
// ---------------------------------------------------------------------------
__global__ void bn_relu_kernel(const float* __restrict__ h2,
                               const float* __restrict__ bnsum,
                               const float* __restrict__ bnsq,
                               const float* __restrict__ g,
                               const float* __restrict__ b,
                               float* __restrict__ h) {
    const float invN = 1.0f / (float)N_NODES;
    int i = blockIdx.x * blockDim.x + threadIdx.x;
    int st = gridDim.x * blockDim.x;
    for (; i < N_NODES * HIDDEN; i += st) {
        int c = i & 127;
        float mu = bnsum[c] * invN;
        float var = bnsq[c] * invN - mu * mu;
        float v = (h2[i] - mu) * rsqrtf(var + BN_EPS) * g[c] + b[c];
        h[i] = fmaxf(v, 0.0f);
    }
}

// ---------------------------------------------------------------------------
// Global mean pool (segment sums). One wave per node, 4 channels/lane.
// ---------------------------------------------------------------------------
__global__ __launch_bounds__(256)
void pool_kernel(const float* __restrict__ h, const int* __restrict__ batch,
                 float* __restrict__ pool, float* __restrict__ cnt) {
    int lane = threadIdx.x & 31;
    int gw = (blockIdx.x * blockDim.x + threadIdx.x) >> 5;
    int nw = (gridDim.x * blockDim.x) >> 5;
    const float4* h4 = (const float4*)h;
    for (int node = gw; node < N_NODES; node += nw) {
        int g = batch[node];
        float4 v = h4[(size_t)node * 32 + lane];
        float* p = pool + (size_t)g * HIDDEN + lane * 4;
        atomicAdd(p + 0, v.x);
        atomicAdd(p + 1, v.y);
        atomicAdd(p + 2, v.z);
        atomicAdd(p + 3, v.w);
        if (lane == 0) atomicAdd(&cnt[g], 1.0f);
    }
}

// ---------------------------------------------------------------------------
// MLP head: 128 -> 64 -> 32 -> 1 with LeakyReLU(0.2). One block per graph.
// ---------------------------------------------------------------------------
__global__ __launch_bounds__(64)
void head_kernel(const float* __restrict__ pool, const float* __restrict__ cnt,
                 const float* __restrict__ W0, const float* __restrict__ B0,
                 const float* __restrict__ W1, const float* __restrict__ B1,
                 const float* __restrict__ W2, const float* __restrict__ B2,
                 float* __restrict__ out) {
    __shared__ float gv[128], l1[64], l2[32];
    int g = blockIdx.x, t = threadIdx.x;
    float c = fmaxf(cnt[g], 1.0f);
    for (int i = t; i < 128; i += 64) gv[i] = pool[(size_t)g * HIDDEN + i] / c;
    __syncthreads();
    {
        float s = B0[t];
        for (int k = 0; k < 128; ++k) s += gv[k] * W0[k * 64 + t];
        l1[t] = s > 0.0f ? s : 0.2f * s;
    }
    __syncthreads();
    if (t < 32) {
        float s = B1[t];
        for (int k = 0; k < 64; ++k) s += l1[k] * W1[k * 32 + t];
        l2[t] = s > 0.0f ? s : 0.2f * s;
    }
    __syncthreads();
    if (t == 0) {
        float s = B2[0];
        for (int k = 0; k < 32; ++k) s += l2[k] * W2[k];
        out[g] = s;
    }
}

// ---------------------------------------------------------------------------
// Host launcher.
// Input order (JAX pytree / sorted-key flattening):
//   0:x  1:edge_index  2:edge_attr  3:batch
//   per conv layer L (base = 4 + 9L):
//     +0 bn_b  +1 bn_g  +2 edge.b  +3 edge.w  +4 eps  +5 nn1.b  +6 nn1.w
//     +7 nn2.b +8 nn2.w
//   40: mlp0.b  41: mlp0.w  42: mlp1.b  43: mlp1.w  44: mlp2.b  45: mlp2.w
// ---------------------------------------------------------------------------
extern "C" void kernel_launch(void* const* d_in, const int* in_sizes, int n_in,
                              void* d_out, int out_size, void* d_ws, size_t ws_size,
                              hipStream_t stream) {
    (void)in_sizes; (void)n_in; (void)out_size; (void)ws_size;

    const float* x = (const float*)d_in[0];
    const int* ei = (const int*)d_in[1];
    const int* src = ei;
    const int* dst = ei + N_EDGES;
    const float* ea = (const float*)d_in[2];
    const int* batch = (const int*)d_in[3];

    auto cp = [&](int L, int slot) { return (const float*)d_in[4 + 9 * L + slot]; };

    // workspace layout
    const size_t NH = (size_t)N_NODES * HIDDEN;
    float* agg = (float*)d_ws;
    float* h2 = agg + NH;
    float* h = h2 + NH;
    float* bnsum = h + NH;                              // 128
    float* bnsq = bnsum + 128;                          // 128
    float* pool = bnsq + 128;                           // 256*128
    float* cnt = pool + (size_t)N_GRAPHS * HIDDEN;      // 256
    __bf16* wf1 = (__bf16*)(cnt + N_GRAPHS);            // 4*4096 bf16 max
    __bf16* wf2 = wf1 + 4 * 4096;                       // 4*4096 bf16

    const float* hin = x;
    int din = 10;

    for (int L = 0; L < 4; ++L) {
        int aggN = N_NODES * din;
        zero_kernel<<<2048, 256, 0, stream>>>(agg, aggN);
        zero_kernel<<<1, 256, 0, stream>>>(bnsum, 256);  // bnsum + bnsq contiguous

        int kch = (din == 10) ? 1 : 4;
        wfrag_kernel<<<kch, 256, 0, stream>>>(cp(L, 6), wf1, din);    // nn1.w
        wfrag_kernel<<<4, 256, 0, stream>>>(cp(L, 8), wf2, HIDDEN);   // nn2.w

        if (din == 10) {
            edge_kernel10<<<(N_EDGES + 255) / 256, 256, 0, stream>>>(
                ea, cp(L, 3), cp(L, 2), hin, src, dst, agg);
        } else {
            edge_kernel128<<<1024, 256, 0, stream>>>(
                ea, cp(L, 3), cp(L, 2), hin, src, dst, agg);
        }

        if (din == 10) {
            mlp_wmma_kernel<10><<<(N_NODES + 127) / 128, 256, 0, stream>>>(
                hin, agg, wf1, cp(L, 5), wf2, cp(L, 7), cp(L, 4),
                h2, bnsum, bnsq);
        } else {
            mlp_wmma_kernel<128><<<(N_NODES + 127) / 128, 256, 0, stream>>>(
                hin, agg, wf1, cp(L, 5), wf2, cp(L, 7), cp(L, 4),
                h2, bnsum, bnsq);
        }

        bn_relu_kernel<<<2048, 256, 0, stream>>>(h2, bnsum, bnsq, cp(L, 1), cp(L, 0), h);

        hin = h;
        din = HIDDEN;
    }

    zero_kernel<<<64, 256, 0, stream>>>(pool, N_GRAPHS * HIDDEN + N_GRAPHS);
    pool_kernel<<<1024, 256, 0, stream>>>(h, batch, pool, cnt);

    head_kernel<<<N_GRAPHS, 64, 0, stream>>>(
        pool, cnt,
        (const float*)d_in[41], (const float*)d_in[40],
        (const float*)d_in[43], (const float*)d_in[42],
        (const float*)d_in[45], (const float*)d_in[44],
        (float*)d_out);
}
